// PointNet_40295383171459
// MI455X (gfx1250) — matrix operand order repaired and checked by reading, not compiled, not run on hardware
//
#include <hip/hip_runtime.h>
#include <hip/hip_bf16.h>

#define N_NODES   100000
#define N_EDGES   1600000
#define NUM_GRAPHS 64

typedef __attribute__((ext_vector_type(16))) _Float16 v16h;
typedef __attribute__((ext_vector_type(8)))  _Float16 v8h;
typedef __attribute__((ext_vector_type(8)))  float    v8f;

__device__ __forceinline__ v8h ld8h(const _Float16* p) {
  return *(const v8h*)p;
}
__device__ __forceinline__ v16h pack16(v8h a, v8h b) {
  v16h r;
#pragma unroll
  for (int e = 0; e < 8; ++e) { r[e] = a[e]; r[e + 8] = b[e]; }
  return r;
}
__device__ __forceinline__ v16h zero16() {
  v16h r;
#pragma unroll
  for (int e = 0; e < 16; ++e) r[e] = (_Float16)0.0f;
  return r;
}
// Native CDNA5 float atomic max (segment-max scatter), device scope.
__device__ __forceinline__ void atomic_max_f32(float* addr, float v) {
  asm volatile("global_atomic_max_num_f32 %0, %1, off scope:SCOPE_DEV"
               :: "v"(addr), "v"(v) : "memory");
}

// ---------------------------------------------------------------------------
// Workspace init: node accumulators = -inf (seg-max identity pre-state),
// stats = 0. Re-run every call (harness does not re-poison).
// ---------------------------------------------------------------------------
__global__ __launch_bounds__(256) void k_init(float* n1, float* n2, float* gp,
                                              float* stats) {
  long long i = (long long)blockIdx.x * 256 + threadIdx.x;
  const float ninf = __int_as_float(0xff800000);
  if (i < (long long)N_NODES * 32) n1[i] = ninf;
  if (i < (long long)N_NODES * 64) n2[i] = ninf;
  if (i < NUM_GRAPHS * 64) gp[i] = ninf;
  if (i < 192) stats[i] = 0.0f;
}

// ---------------------------------------------------------------------------
// Layer 1: edge MLP  cat([pos_j, pos_j-pos_i]) (K=6 pad 32) -> 32 -> ReLU -> 32
// One wave per 16-edge tile; 4x v_wmma_f32_16x16x32_f16 per tile.
// ---------------------------------------------------------------------------
__global__ __launch_bounds__(256) void k_layer1(
    const float* __restrict__ pos, const int* __restrict__ ei,
    const float* __restrict__ W1, const float* __restrict__ B1,
    const float* __restrict__ W2, const float* __restrict__ B2,
    float* __restrict__ nodeMax) {
  __shared__ alignas(16) _Float16 sW1T[32][32];   // [n][k], k>=6 zero-padded
  __shared__ alignas(16) _Float16 sW2T[32][32];   // [n][k]
  __shared__ float sB1[32], sB2[32];
  __shared__ int sDst[8][16];
  __shared__ alignas(16) _Float16 sHid[8][16][32]; // per-wave hidden staging

  const int tid = threadIdx.x;
  for (int i = tid; i < 32 * 32; i += 256) {
    int n = i >> 5, k = i & 31;
    sW1T[n][k] = (_Float16)(k < 6 ? W1[k * 32 + n] : 0.0f);
    sW2T[n][k] = (_Float16)W2[k * 32 + n];
  }
  if (tid < 32) { sB1[tid] = B1[tid]; sB2[tid] = B2[tid]; }
  __syncthreads();

  const int wave = tid >> 5, lane = tid & 31;
  const int lo = lane & 15;
  const bool hi = lane >= 16;
  const long long eBase = ((long long)blockIdx.x * 8 + wave) * 16;
  if (eBase >= (long long)N_EDGES) return;   // uniform per wave

  // Build A (16x32 f16): lanes<16 hold K=0..15 (feats at 0..5, rest zero),
  // lanes>=16 hold K=16..31 (all zero for K=6).
  v16h a = zero16();
  if (!hi) {
    long long e = eBase + lo;
    int s = ei[e];
    int d = ei[N_EDGES + e];
    sDst[wave][lo] = d;
    float pjx = pos[3 * s], pjy = pos[3 * s + 1], pjz = pos[3 * s + 2];
    float pix = pos[3 * d], piy = pos[3 * d + 1], piz = pos[3 * d + 2];
    a[0] = (_Float16)pjx; a[1] = (_Float16)pjy; a[2] = (_Float16)pjz;
    a[3] = (_Float16)(pjx - pix); a[4] = (_Float16)(pjy - piy);
    a[5] = (_Float16)(pjz - piz);
  }

  // Linear 1 (6->32): two 16-col tiles, bias in C.
  v8f acc[2];
#pragma unroll
  for (int ct = 0; ct < 2; ++ct) {
    const int n = ct * 16 + lo;
    const int kb = hi ? 16 : 0;
    v16h b = pack16(ld8h(&sW1T[n][kb]), ld8h(&sW1T[n][kb + 8]));
    v8f c;
    float bias = sB1[n];
#pragma unroll
    for (int r = 0; r < 8; ++r) c[r] = bias;
    acc[ct] = __builtin_amdgcn_wmma_f32_16x16x32_f16(
        false, a, false, b, (short)0, c, false, false);
  }
  // ReLU + stage hidden [16 edges][32 ch] f16 (D layout -> row major via LDS)
#pragma unroll
  for (int ct = 0; ct < 2; ++ct) {
#pragma unroll
    for (int r = 0; r < 8; ++r) {
      int M = r + (hi ? 8 : 0);
      float v = acc[ct][r];
      sHid[wave][M][ct * 16 + lo] = (_Float16)(v > 0.0f ? v : 0.0f);
    }
  }
  // Reload in A layout (per-wave LDS, DS ops in-order within wave)
  const _Float16* row = &sHid[wave][lo][0];
  {
    const int kb = hi ? 16 : 0;
    a = pack16(ld8h(row + kb), ld8h(row + kb + 8));
  }
  // Linear 2 (32->32) + scatter-max to dst nodes.
#pragma unroll
  for (int ct = 0; ct < 2; ++ct) {
    const int n = ct * 16 + lo;
    const int kb = hi ? 16 : 0;
    v16h b = pack16(ld8h(&sW2T[n][kb]), ld8h(&sW2T[n][kb + 8]));
    v8f c;
    float bias = sB2[n];
#pragma unroll
    for (int r = 0; r < 8; ++r) c[r] = bias;
    c = __builtin_amdgcn_wmma_f32_16x16x32_f16(
        false, a, false, b, (short)0, c, false, false);
#pragma unroll
    for (int r = 0; r < 8; ++r) {
      int M = r + (hi ? 8 : 0);
      int d = sDst[wave][M];
      atomic_max_f32(nodeMax + (long long)d * 32 + n, c[r]);
    }
  }
}

// ---------------------------------------------------------------------------
// Finalize seg_max0 identity (-inf -> 0) + per-channel sum / sumsq partials
// (LDS ds_add_f32 then one global atomicAdd per channel per block).
// ---------------------------------------------------------------------------
__global__ __launch_bounds__(256) void k_finalize_stats(float* node, int nElem,
                                                        int C, float* sum,
                                                        float* sumsq) {
  __shared__ float lsum[64], lsq[64];
  int t = threadIdx.x;
  if (t < C) { lsum[t] = 0.0f; lsq[t] = 0.0f; }
  __syncthreads();
  int i = blockIdx.x * 256 + t;
  if (i < nElem) {
    float v = node[i];
    if (v < -1e30f) v = 0.0f;     // empty segment -> 0
    node[i] = v;
    atomicAdd(&lsum[t % C], v);   // ds_add_f32
    atomicAdd(&lsq[t % C], v * v);
  }
  __syncthreads();
  if (t < C) { atomicAdd(&sum[t], lsum[t]); atomicAdd(&sumsq[t], lsq[t]); }
}

// coef[c] = gamma*rsqrt(var+eps); coef[C+c] = beta - mu*coef[c]
__global__ void k_bn_coef(const float* sum, const float* sumsq,
                          const float* gamma, const float* beta, float* coef,
                          int C, float invN) {
  int c = threadIdx.x;
  if (c >= C) return;
  float mu = sum[c] * invN;
  float var = sumsq[c] * invN - mu * mu;
  float s = rsqrtf(var + 1e-5f) * gamma[c];
  coef[c] = s;
  coef[C + c] = beta[c] - mu * s;
}

// Apply BN affine + ReLU, emit f16 node features for layer-2 gather.
__global__ __launch_bounds__(256) void k_bn_relu_f16(const float* node,
                                                     const float* coef,
                                                     _Float16* out, int nElem) {
  int i = blockIdx.x * 256 + threadIdx.x;
  if (i >= nElem) return;
  int c = i & 31;
  float v = node[i] * coef[c] + coef[32 + c];
  out[i] = (_Float16)(v > 0.0f ? v : 0.0f);
}

// ---------------------------------------------------------------------------
// Layer 2: cat([h_j(32), pos_j-pos_i]) (K=35 pad 64) -> 64 -> ReLU -> 64.
// 16x v_wmma_f32_16x16x32_f16 per 16-edge tile.
// ---------------------------------------------------------------------------
__global__ __launch_bounds__(256) void k_layer2(
    const float* __restrict__ pos, const int* __restrict__ ei,
    const _Float16* __restrict__ hIn,
    const float* __restrict__ W1, const float* __restrict__ B1,
    const float* __restrict__ W2, const float* __restrict__ B2,
    float* __restrict__ nodeMax) {
  __shared__ alignas(16) _Float16 sW1T[64][64];   // [n][k], k>=35 zero
  __shared__ alignas(16) _Float16 sW2T[64][64];   // [n][k]
  __shared__ float sB1[64], sB2[64];
  __shared__ int sSrc[8][16], sDst[8][16];
  __shared__ alignas(16) _Float16 sHid[8][16][64];

  const int tid = threadIdx.x;
  for (int i = tid; i < 64 * 64; i += 256) {
    int n = i >> 6, k = i & 63;
    sW1T[n][k] = (_Float16)(k < 35 ? W1[k * 64 + n] : 0.0f);
    sW2T[n][k] = (_Float16)W2[k * 64 + n];
  }
  if (tid < 64) { sB1[tid] = B1[tid]; sB2[tid] = B2[tid]; }
  __syncthreads();

  const int wave = tid >> 5, lane = tid & 31;
  const int lo = lane & 15;
  const bool hi = lane >= 16;
  const long long eBase = ((long long)blockIdx.x * 8 + wave) * 16;
  if (eBase >= (long long)N_EDGES) return;

  float rx = 0.f, ry = 0.f, rz = 0.f;
  if (!hi) {
    long long e = eBase + lo;
    int s = ei[e];
    int d = ei[N_EDGES + e];
    sSrc[wave][lo] = s;
    sDst[wave][lo] = d;
    rx = pos[3 * s] - pos[3 * d];
    ry = pos[3 * s + 1] - pos[3 * d + 1];
    rz = pos[3 * s + 2] - pos[3 * d + 2];
  }
  // Gather h[src] straight into A layout: lane pair (l, l+16) splits the 32 ch.
  int s = sSrc[wave][lo];                 // DS in-order within wave
  const _Float16* hp = hIn + (long long)s * 32;
  const int kb = hi ? 16 : 0;
  v16h a0 = pack16(ld8h(hp + kb), ld8h(hp + kb + 8));    // K=0..31 = h
  v16h a1 = zero16();                                    // K=32..34 = rel pos
  if (!hi) { a1[0] = (_Float16)rx; a1[1] = (_Float16)ry; a1[2] = (_Float16)rz; }

  // Linear 1 (35->64): 4 col tiles x 2 K-steps.
  v8f acc[4];
#pragma unroll
  for (int ct = 0; ct < 4; ++ct) {
    const int n = ct * 16 + lo;
    const int kbB = hi ? 16 : 0;
    v16h b0 = pack16(ld8h(&sW1T[n][kbB]), ld8h(&sW1T[n][kbB + 8]));
    v16h b1 = pack16(ld8h(&sW1T[n][32 + kbB]), ld8h(&sW1T[n][32 + kbB + 8]));
    v8f c;
    float bias = sB1[n];
#pragma unroll
    for (int r = 0; r < 8; ++r) c[r] = bias;
    c = __builtin_amdgcn_wmma_f32_16x16x32_f16(false, a0, false, b0, (short)0, c, false, false);
    c = __builtin_amdgcn_wmma_f32_16x16x32_f16(false, a1, false, b1, (short)0, c, false, false);
    acc[ct] = c;
  }
  // ReLU + stage hidden [16][64]
#pragma unroll
  for (int ct = 0; ct < 4; ++ct) {
#pragma unroll
    for (int r = 0; r < 8; ++r) {
      int M = r + (hi ? 8 : 0);
      float v = acc[ct][r];
      sHid[wave][M][ct * 16 + lo] = (_Float16)(v > 0.0f ? v : 0.0f);
    }
  }
  const _Float16* row = &sHid[wave][lo][0];
  v16h h0 = pack16(ld8h(row + kb), ld8h(row + kb + 8));            // K=0..31
  v16h h1 = pack16(ld8h(row + 32 + kb), ld8h(row + 32 + kb + 8));  // K=32..63

  // Linear 2 (64->64) + scatter-max.
#pragma unroll
  for (int ct = 0; ct < 4; ++ct) {
    const int n = ct * 16 + lo;
    const int kbB = hi ? 16 : 0;
    v16h b0 = pack16(ld8h(&sW2T[n][kbB]), ld8h(&sW2T[n][kbB + 8]));
    v16h b1 = pack16(ld8h(&sW2T[n][32 + kbB]), ld8h(&sW2T[n][32 + kbB + 8]));
    v8f c;
    float bias = sB2[n];
#pragma unroll
    for (int r = 0; r < 8; ++r) c[r] = bias;
    c = __builtin_amdgcn_wmma_f32_16x16x32_f16(false, h0, false, b0, (short)0, c, false, false);
    c = __builtin_amdgcn_wmma_f32_16x16x32_f16(false, h1, false, b1, (short)0, c, false, false);
#pragma unroll
    for (int r = 0; r < 8; ++r) {
      int M = r + (hi ? 8 : 0);
      int d = sDst[wave][M];
      atomic_max_f32(nodeMax + (long long)d * 64 + n, c[r]);
    }
  }
}

// BN2 affine + ReLU + graph-level segment max (batch is sorted -> L2 friendly)
__global__ __launch_bounds__(256) void k_pool(const float* node2,
                                              const float* coef,
                                              const int* batch, float* gp,
                                              int nElem) {
  int i = blockIdx.x * 256 + threadIdx.x;
  if (i >= nElem) return;
  int c = i & 63;
  int nidx = i >> 6;
  float v = node2[i] * coef[c] + coef[64 + c];
  v = v > 0.0f ? v : 0.0f;
  atomic_max_f32(gp + (long long)batch[nidx] * 64 + c, v);
}

// Final head: [64,64] @ [64,2] + bc  (tiny; one block)
__global__ void k_head(const float* gp, const float* Wc, const float* bc,
                       float* out) {
  int t = threadIdx.x;                       // 128 threads
  int g = t >> 1, o = t & 1;
  float acc = bc[o];
#pragma unroll 8
  for (int c = 0; c < 64; ++c) {
    float v = gp[g * 64 + c];
    if (v < -1e30f) v = 0.0f;                // empty graph -> 0
    acc += v * Wc[c * 2 + o];
  }
  out[t] = acc;
}

// ---------------------------------------------------------------------------
extern "C" void kernel_launch(void* const* d_in, const int* in_sizes, int n_in,
                              void* d_out, int out_size, void* d_ws,
                              size_t ws_size, hipStream_t stream) {
  const float* pos   = (const float*)d_in[0];
  const int*   ei    = (const int*)d_in[1];
  const int*   batch = (const int*)d_in[2];
  const float* W1a = (const float*)d_in[3];
  const float* b1a = (const float*)d_in[4];
  const float* W2a = (const float*)d_in[5];
  const float* b2a = (const float*)d_in[6];
  const float* g1    = (const float*)d_in[7];
  const float* beta1 = (const float*)d_in[8];
  const float* W1b = (const float*)d_in[9];
  const float* b1b = (const float*)d_in[10];
  const float* W2b = (const float*)d_in[11];
  const float* b2b = (const float*)d_in[12];
  const float* g2    = (const float*)d_in[13];
  const float* beta2 = (const float*)d_in[14];
  const float* Wc = (const float*)d_in[15];
  const float* bc = (const float*)d_in[16];
  float* out = (float*)d_out;

  // Workspace layout (bytes)
  char* ws = (char*)d_ws;
  float* node1 = (float*)(ws);                         // 100000*32 f32
  float* node2 = (float*)(ws + 12800000);              // 100000*64 f32
  float* gpool = (float*)(ws + 38400000);              // 64*64 f32
  float* stats = (float*)(ws + 38416384);              // sum1(32) sq1(32) sum2(64) sq2(64)
  float* coef1 = (float*)(ws + 38417152);              // 64
  float* coef2 = coef1 + 64;                           // 128
  _Float16* h1 = (_Float16*)(ws + 38417920);           // 100000*32 f16

  const int nE1 = N_NODES * 32;                        // 3,200,000
  const int nE2 = N_NODES * 64;                        // 6,400,000
  const int edgeBlocks = N_EDGES / (8 * 16);           // 12,500
  const float invN = 1.0f / (float)N_NODES;

  k_init<<<25000, 256, 0, stream>>>(node1, node2, gpool, stats);
  k_layer1<<<edgeBlocks, 256, 0, stream>>>(pos, ei, W1a, b1a, W2a, b2a, node1);
  k_finalize_stats<<<nE1 / 256, 256, 0, stream>>>(node1, nE1, 32, stats, stats + 32);
  k_bn_coef<<<1, 64, 0, stream>>>(stats, stats + 32, g1, beta1, coef1, 32, invN);
  k_bn_relu_f16<<<nE1 / 256, 256, 0, stream>>>(node1, coef1, h1, nE1);
  k_layer2<<<edgeBlocks, 256, 0, stream>>>(pos, ei, h1, W1b, b1b, W2b, b2b, node2);
  k_finalize_stats<<<nE2 / 256, 256, 0, stream>>>(node2, nE2, 64, stats + 64, stats + 128);
  k_bn_coef<<<1, 64, 0, stream>>>(stats + 64, stats + 128, g2, beta2, coef2, 64, invN);
  k_pool<<<nE2 / 256, 256, 0, stream>>>(node2, coef2, batch, gpool, nE2);
  k_head<<<1, 128, 0, stream>>>(gpool, Wc, bc, out);
}
